// SSDTrainingModel_34127810134544
// MI455X (gfx1250) — compile-verified, matrix operand order checked by reading, not run on hardware
//
#include <hip/hip_runtime.h>
#include <hip/hip_bf16.h>
#include <float.h>

// ---- problem constants (from reference) ----
#define BB 64
#define AA 24564
#define LL 20
#define CC 21
#define POS_THR 0.5f
#define NEG_THR 0.3f
#define SCORE_THR 0.01f
#define NMS_THR 0.45f
#define MAX_DET 100
#define K_PRE 400
#define CVc 0.1f
#define SVc 0.2f
#define ALPHAc 0.25f
#define NEG_INF (-1e30f)

#define TILE_AN 256                    // anchors staged per NMS tile
#define TILE_FLOATS (TILE_AN * 25)     // 6400 f32 = 25600 B
#define TILE_B128 (TILE_FLOATS / 4)    // 1600 x 16B
#define TILE_E8 (TILE_FLOATS / 2)      // 3200 x 8B elements for TDM

typedef __attribute__((ext_vector_type(2)))  float        v2f;
typedef __attribute__((ext_vector_type(8)))  float        v8f;
typedef __attribute__((ext_vector_type(16))) _Float16     v16h;
typedef __attribute__((ext_vector_type(4)))  unsigned int v4u;
typedef __attribute__((ext_vector_type(8)))  int          v8i;
typedef __attribute__((ext_vector_type(4)))  int          v4i;

#define HAS_TDM (__has_builtin(__builtin_amdgcn_tensor_load_to_lds) && \
                 __has_builtin(__builtin_amdgcn_s_wait_tensorcnt))

// ---------------------------------------------------------------------------
// Cross-lane pair reduction on the matrix pipe: ones-matrix WMMA collapses the
// wave's 32 partials to 16 column sums (lane n of D[0] = x[n] + x[n+16]).
// EXEC must be all ones at the call site (no early returns in callers).
// ---------------------------------------------------------------------------
__device__ __forceinline__ float wmma_pair_reduce(float x) {
#if __has_builtin(__builtin_amdgcn_wmma_f32_16x16x4_f32)
  v2f a; a[0] = 1.0f; a[1] = 1.0f;
  v2f b; b[0] = x;    b[1] = 0.0f;
  v8f c = {};
  v8f d = __builtin_amdgcn_wmma_f32_16x16x4_f32(false, a, false, b,
                                                (short)0, c, false, false);
  return d[0];
#else
  v16h a; v16h b;
#pragma unroll
  for (int i = 0; i < 16; ++i) { a[i] = (_Float16)1.0f; b[i] = (_Float16)0.0f; }
  b[0] = (_Float16)x;
  v8f c = {};
  v8f d = __builtin_amdgcn_wmma_f32_16x16x32_f16(false, a, false, b,
                                                 (short)0, c, false, false);
  return d[0];
#endif
}

__device__ __forceinline__ float iou_box(float ax0, float ay0, float ax1, float ay1,
                                         float bx0, float by0, float bx1, float by1) {
  float ix0 = fmaxf(ax0, bx0), iy0 = fmaxf(ay0, by0);
  float ix1 = fminf(ax1, bx1), iy1 = fminf(ay1, by1);
  float iw = fmaxf(ix1 - ix0, 0.0f), ih = fmaxf(iy1 - iy0, 0.0f);
  float inter = iw * ih;
  float ua = (ax1 - ax0) * (ay1 - ay0) + (bx1 - bx0) * (by1 - by0) - inter;
  return inter / fmaxf(ua, 1e-8f);
}

// ---------------------------------------------------------------------------
// Stage one tile (TILE_B128 x 16B) of a batch row into LDS.
// Preferred path: Tensor Data Mover 1-D descriptor copy of TILE_E8 8-byte
// elements; tensor_dim0 = elements remaining in the row so tail reads OOB
// return zero.  One wave issues the DMA (TDM ignores EXEC), waits TENSORcnt,
// then the workgroup barrier publishes the LDS tile to all waves.
// Fallback: coalesced float4 global->LDS copy with tail clamping.
// ---------------------------------------------------------------------------
__device__ __forceinline__ void stage_tile(const float* __restrict__ rowBase,
                                           size_t rowBytes, size_t byte0,
                                           float* __restrict__ tile, int tid) {
#if HAS_TDM
  if (tid < 32) {
    unsigned long long ga = (unsigned long long)(uintptr_t)((const char*)rowBase + byte0);
    unsigned ldsAddr = (unsigned)(uintptr_t)tile;   // generic low 32 bits = LDS offset
    size_t remain8s = (rowBytes - byte0) >> 3;
    unsigned t0 = (unsigned)(remain8s > 0xFFFFFFFFull ? 0xFFFFFFFFull : remain8s);

    v4u g0;
    g0[0] = 1u;                                          // count=1, user descriptor
    g0[1] = ldsAddr;                                     // lds_addr [63:32]
    g0[2] = (unsigned)(ga & 0xFFFFFFFFull);              // global_addr low
    g0[3] = (unsigned)((ga >> 32) & 0x1FFFFFFull) | (2u << 30);  // addr hi + type=2

    v8i g1;
    g1[0] = (int)(3u << 16);                             // wg_mask=0, data_size=3 (8B)
    g1[1] = (int)((t0 & 0xFFFFu) << 16);                 // tensor_dim0 low16 @ [63:48]
    g1[2] = (int)(((t0 >> 16) & 0xFFFFu) | (1u << 16));  // tensor_dim0 hi16, tensor_dim1=1
    g1[3] = (int)((unsigned)TILE_E8 << 16);              // tensor_dim1 hi=0, tile_dim0
    g1[4] = 0;                                           // tile_dim1=0 (1-D), tile_dim2=0
    g1[5] = (int)t0;                                     // tensor_dim0_stride low32
    g1[6] = 0;                                           // stride hi16, dim1_stride lo16
    g1[7] = 0;

    v4i gz = {0, 0, 0, 0};
#if defined(__clang_major__) && (__clang_major__ >= 23)
    v8i gz8 = {0, 0, 0, 0, 0, 0, 0, 0};
    __builtin_amdgcn_tensor_load_to_lds(g0, g1, gz, gz, gz8, 0);
#else
    __builtin_amdgcn_tensor_load_to_lds(g0, g1, gz, gz, 0);
#endif
    __builtin_amdgcn_s_wait_tensorcnt((short)0);
  }
#else
  for (int i = tid; i < TILE_B128; i += 256) {
    size_t off = byte0 + (size_t)i * 16;
    if (off + 16 > rowBytes) off = rowBytes - 16;       // tail clamp (dead anchors)
    __builtin_prefetch((const char*)rowBase + off + 4096, 0, 0);
    float4 v = *(const float4*)((const char*)rowBase + off);
    *(float4*)((char*)tile + (size_t)i * 16) = v;
  }
#endif
  __syncthreads();
}

// ---------------------------------------------------------------------------
// K0: zero workspace (best-anchor keys + loss accumulators) every call.
// ---------------------------------------------------------------------------
__global__ void k_init(unsigned long long* bestKey, float* lossAcc) {
  int tid = threadIdx.x;
  for (int i = tid; i < BB * LL; i += blockDim.x) bestKey[i] = 0ull;
  if (tid < 2) lossAcc[tid] = 0.0f;
}

// ---------------------------------------------------------------------------
// K1: per-(b,l) argmax over anchors of IoU via packed u64 atomic-max.
// key = (iou_bits << 32) | (~anchor)  -> max key == max iou, ties -> low anchor.
// ---------------------------------------------------------------------------
__global__ void k_best_anchor(const float* __restrict__ anchors,
                              const float* __restrict__ gt,
                              unsigned long long* __restrict__ bestKey) {
  const int b = blockIdx.y;
  const int tid = threadIdx.x;
  const int a = blockIdx.x * 256 + tid;

  __shared__ float gbox[LL][4];
  __shared__ int   gvalid[LL];
  __shared__ unsigned long long skey[LL];
  if (tid < LL) {
    const float* g = gt + ((size_t)b * LL + tid) * 5;
    int cls = (int)g[0];
    gvalid[tid] = (cls > 0);
    gbox[tid][0] = g[1]; gbox[tid][1] = g[2];
    gbox[tid][2] = g[3]; gbox[tid][3] = g[4];
    skey[tid] = 0ull;
  }
  __syncthreads();

  if (a < AA) {
    float ax0 = anchors[a * 4 + 0], ay0 = anchors[a * 4 + 1];
    float ax1 = anchors[a * 4 + 2], ay1 = anchors[a * 4 + 3];
#pragma unroll
    for (int l = 0; l < LL; ++l) {
      if (gvalid[l]) {
        float iou = iou_box(gbox[l][0], gbox[l][1], gbox[l][2], gbox[l][3],
                            ax0, ay0, ax1, ay1);           // iou >= 0
        unsigned long long key =
            ((unsigned long long)__float_as_uint(iou) << 32) |
            (unsigned long long)(0xFFFFFFFFu - (unsigned)a);
        atomicMax(&skey[l], key);                          // ds_max_u64
      }
    }
  }
  __syncthreads();
  if (tid < LL && gvalid[tid]) atomicMax(&bestKey[b * LL + tid], skey[tid]);
}

// ---------------------------------------------------------------------------
// K2: fused match + focal/smooth-L1 loss.  One thread per (b, anchor).
// No early returns: tail threads clamp index and contribute zero so EXEC is
// all-ones at the WMMA reduction.
// ---------------------------------------------------------------------------
__global__ void k_loss(const float* __restrict__ anchors,
                       const float* __restrict__ gt,
                       const float* __restrict__ y_pred,
                       const unsigned long long* __restrict__ bestKey,
                       float* __restrict__ lossAcc) {
  const int b = blockIdx.y;
  const int tid = threadIdx.x;
  const int a = blockIdx.x * 256 + tid;
  const bool live = (a < AA);
  const int ac = live ? a : (AA - 1);

  __shared__ float gbox[LL][4];
  __shared__ int   gcls[LL];
  __shared__ int   gvalid[LL];
  __shared__ int   fAnchor[LL];
  if (tid < LL) {
    const float* g = gt + ((size_t)b * LL + tid) * 5;
    int cls = (int)g[0];
    gcls[tid] = cls;
    gvalid[tid] = (cls > 0);
    gbox[tid][0] = g[1]; gbox[tid][1] = g[2];
    gbox[tid][2] = g[3]; gbox[tid][3] = g[4];
    unsigned long long k = bestKey[b * LL + tid];
    fAnchor[tid] = (int)(0xFFFFFFFFu - (unsigned)(k & 0xFFFFFFFFull));
  }
  __syncthreads();

  float ax0 = anchors[ac * 4 + 0], ay0 = anchors[ac * 4 + 1];
  float ax1 = anchors[ac * 4 + 2], ay1 = anchors[ac * 4 + 3];
  float acx = 0.5f * (ax0 + ax1), acy = 0.5f * (ay0 + ay1);
  float aw = ax1 - ax0, ah = ay1 - ay0;

  float bIou = -1.0f; int bIdx = 0;
#pragma unroll
  for (int l = 0; l < LL; ++l) {
    float iou = iou_box(gbox[l][0], gbox[l][1], gbox[l][2], gbox[l][3],
                        ax0, ay0, ax1, ay1);
    float val = gvalid[l] ? iou : -1.0f;
    if (val > bIou) { bIou = val; bIdx = l; }
  }
  int forced = -1;
#pragma unroll
  for (int l = 0; l < LL; ++l)
    if (gvalid[l] && fAnchor[l] == ac) forced = l;

  int assigned = (forced >= 0) ? forced : bIdx;
  bool pos = (forced >= 0) || (bIou >= POS_THR);
  bool neg = (!pos) && (bIou < NEG_THR);

  const float* yl = y_pred + ((size_t)b * AA + ac) * (CC + 4);
  float lg[CC];
#pragma unroll
  for (int c = 0; c < CC; ++c) lg[c] = yl[c];
  float m = lg[0];
#pragma unroll
  for (int c = 1; c < CC; ++c) m = fmaxf(m, lg[c]);
  float s = 0.0f;
#pragma unroll
  for (int c = 0; c < CC; ++c) s += __expf(lg[c] - m);
  float logs = __logf(s);

  int cc = pos ? gcls[assigned] : 0;
  float logp = lg[cc] - m - logs;
  float p = __expf(logp);
  float clsl = (pos || neg) ? (-ALPHAc * (1.0f - p) * (1.0f - p) * logp) : 0.0f;

  float locl = 0.0f;
  if (pos) {
    float gx0 = gbox[assigned][0], gy0 = gbox[assigned][1];
    float gx1 = gbox[assigned][2], gy1 = gbox[assigned][3];
    float gcx = 0.5f * (gx0 + gx1), gcy = 0.5f * (gy0 + gy1);
    float gw = fmaxf(gx1 - gx0, 1e-8f), gh = fmaxf(gy1 - gy0, 1e-8f);
    float off[4];
    off[0] = (gcx - acx) / (aw * CVc);
    off[1] = (gcy - acy) / (ah * CVc);
    off[2] = __logf(gw / aw) / SVc;
    off[3] = __logf(gh / ah) / SVc;
#pragma unroll
    for (int k = 0; k < 4; ++k) {
      float d = fabsf(yl[CC + k] - off[k]);
      locl += (d < 1.0f) ? 0.5f * d * d : d - 0.5f;
    }
  }

  float partial = live ? (clsl + locl) : 0.0f;
  float pcount  = (live && pos) ? 1.0f : 0.0f;

  float rl = wmma_pair_reduce(partial);   // matrix-pipe 32 -> 16
  float rp = wmma_pair_reduce(pcount);

  __shared__ float red0[128];
  __shared__ float red1[128];
  int lane = tid & 31, w = tid >> 5;
  if (lane < 16) { red0[w * 16 + lane] = rl; red1[w * 16 + lane] = rp; }
  __syncthreads();
  for (int st = 64; st > 0; st >>= 1) {
    if (tid < st) { red0[tid] += red0[tid + st]; red1[tid] += red1[tid + st]; }
    __syncthreads();
  }
  if (tid == 0) {
    atomicAdd(&lossAcc[0], red0[0]);
    atomicAdd(&lossAcc[1], red1[0]);
  }
}

__global__ void k_finalize(const float* __restrict__ lossAcc,
                           float* __restrict__ out) {
  out[0] = lossAcc[0] / fmaxf(lossAcc[1], 1.0f);
}

// ---------------------------------------------------------------------------
// K3: per-batch top-K_PRE selection (fp32-exponent histogram) + NMS in LDS.
// y_pred tiles are DMA'd into LDS by the TDM; stride-25 LDS rows are
// bank-conflict-free (gcd(25,64)=1).
// ---------------------------------------------------------------------------
__device__ __forceinline__ void softmax_probs(const float* __restrict__ row,
                                              float* p) {
  float lg[CC];
#pragma unroll
  for (int c = 0; c < CC; ++c) lg[c] = row[c];
  float m = lg[0];
#pragma unroll
  for (int c = 1; c < CC; ++c) m = fmaxf(m, lg[c]);
  float s = 0.0f;
#pragma unroll
  for (int c = 0; c < CC; ++c) { p[c] = __expf(lg[c] - m); s += p[c]; }
  float inv = 1.0f / s;
#pragma unroll
  for (int c = 0; c < CC; ++c) p[c] *= inv;
}

__global__ void k_nms(const float* __restrict__ anchors,
                      const float* __restrict__ y_pred,
                      float* __restrict__ det_boxes,
                      float* __restrict__ det_scores,
                      float* __restrict__ det_classes) {
  const int b = blockIdx.x;
  const int tid = threadIdx.x;
  const float* yp = y_pred + (size_t)b * AA * (CC + 4);
  const size_t rowBytes = (size_t)AA * (CC + 4) * sizeof(float);

  __shared__ float    tile[TILE_FLOATS];     // 25.6 KB staging buffer
  __shared__ unsigned hist[128];
  __shared__ int      cutB;
  __shared__ unsigned nHighS, remS, cntHigh, cntCut, nCandS;
  __shared__ float    cscore[K_PRE];
  __shared__ unsigned cinfo[K_PRE];
  __shared__ float    cbox[K_PRE][4];
  __shared__ float    rmax[256];
  __shared__ int      rarg[256];
  __shared__ float    wsc; __shared__ int warg;

  for (int i = tid; i < 128; i += 256) hist[i] = 0u;
  __syncthreads();

  // Pass 1: histogram of score exponents (scores in (0,1] -> monotone bits)
  for (int t0 = 0; t0 < AA; t0 += TILE_AN) {
    stage_tile(yp, rowBytes, (size_t)t0 * (CC + 4) * sizeof(float), tile, tid);
    int a = t0 + tid;
    if (a < AA) {
      float p[CC];
      softmax_probs(&tile[tid * (CC + 4)], p);
#pragma unroll
      for (int c = 1; c < CC; ++c) {
        if (p[c] > SCORE_THR) {
          unsigned bkt = __float_as_uint(p[c]) >> 23;
          if (bkt > 127u) bkt = 127u;
          atomicAdd(&hist[bkt], 1u);
        }
      }
    }
    __syncthreads();   // tile consumed before next DMA overwrites it
  }

  if (tid == 0) {
    unsigned acc = 0; int cb = -1; unsigned rm = 0;
    for (int i = 127; i >= 0; --i) {
      if (acc + hist[i] >= (unsigned)K_PRE) { cb = i; rm = K_PRE - acc; break; }
      acc += hist[i];
    }
    cutB = cb; nHighS = acc; remS = rm; cntHigh = 0; cntCut = 0;
  }
  for (int i = tid; i < K_PRE; i += 256) {
    cscore[i] = NEG_INF; cinfo[i] = 0u;
    cbox[i][0] = cbox[i][1] = cbox[i][2] = cbox[i][3] = 0.0f;
  }
  __syncthreads();

  // Pass 2: compact candidates above / at the cutoff bucket into LDS
  const int cb = cutB; const unsigned nH = nHighS, rm = remS;
  for (int t0 = 0; t0 < AA; t0 += TILE_AN) {
    stage_tile(yp, rowBytes, (size_t)t0 * (CC + 4) * sizeof(float), tile, tid);
    int a = t0 + tid;
    if (a < AA) {
      float p[CC];
      softmax_probs(&tile[tid * (CC + 4)], p);
#pragma unroll
      for (int c = 1; c < CC; ++c) {
        if (p[c] > SCORE_THR) {
          unsigned bkt = __float_as_uint(p[c]) >> 23;
          if (bkt > 127u) bkt = 127u;
          if ((int)bkt > cb) {
            unsigned idx = atomicAdd(&cntHigh, 1u);      // < nH <= K_PRE
            cscore[idx] = p[c]; cinfo[idx] = ((unsigned)a << 5) | (unsigned)c;
          } else if ((int)bkt == cb) {
            unsigned j = atomicAdd(&cntCut, 1u);
            if (j < rm) {
              unsigned idx = nH + j;
              cscore[idx] = p[c]; cinfo[idx] = ((unsigned)a << 5) | (unsigned)c;
            }
          }
        }
      }
    }
    __syncthreads();
  }
  if (tid == 0) {
    unsigned extra = cntCut < remS ? cntCut : remS;
    nCandS = nHighS + extra;
  }
  __syncthreads();

  // Decode boxes for the <=400 surviving candidates (scattered, direct reads)
  const unsigned nc = nCandS;
  for (unsigned i = tid; i < nc; i += 256) {
    unsigned a = cinfo[i] >> 5;
    const float* o = yp + (size_t)a * (CC + 4) + CC;
    float ax0 = anchors[a * 4 + 0], ay0 = anchors[a * 4 + 1];
    float ax1 = anchors[a * 4 + 2], ay1 = anchors[a * 4 + 3];
    float acx = 0.5f * (ax0 + ax1), acy = 0.5f * (ay0 + ay1);
    float aw = ax1 - ax0, ah = ay1 - ay0;
    float cx = o[0] * CVc * aw + acx;
    float cy = o[1] * CVc * ah + acy;
    float w = __expf(o[2] * SVc) * aw;
    float h = __expf(o[3] * SVc) * ah;
    cbox[i][0] = cx - 0.5f * w; cbox[i][1] = cy - 0.5f * h;
    cbox[i][2] = cx + 0.5f * w; cbox[i][3] = cy + 0.5f * h;
  }
  __syncthreads();

  // Sequential NMS rounds, parallel argmax + parallel suppression per round
  float* ob = det_boxes   + (size_t)b * MAX_DET * 4;
  float* os = det_scores  + (size_t)b * MAX_DET;
  float* oc = det_classes + (size_t)b * MAX_DET;

  for (int r = 0; r < MAX_DET; ++r) {
    float lm = -FLT_MAX; int la = -1;
    for (int i = tid; i < K_PRE; i += 256)
      if (cscore[i] > lm) { lm = cscore[i]; la = i; }
    rmax[tid] = lm; rarg[tid] = la;
    __syncthreads();
    for (int st = 128; st > 0; st >>= 1) {
      if (tid < st && rmax[tid + st] > rmax[tid]) {
        rmax[tid] = rmax[tid + st]; rarg[tid] = rarg[tid + st];
      }
      __syncthreads();
    }
    if (tid == 0) { wsc = rmax[0]; warg = rarg[0]; }
    __syncthreads();

    const int wi = warg; const float si = wsc;
    const bool ok = (wi >= 0) && (si > -1e29f);
    float wx0 = 0, wy0 = 0, wx1 = 0, wy1 = 0;
    if (ok) { wx0 = cbox[wi][0]; wy0 = cbox[wi][1]; wx1 = cbox[wi][2]; wy1 = cbox[wi][3]; }

    if (tid == 0) {
      ob[r * 4 + 0] = ok ? wx0 : 0.0f; ob[r * 4 + 1] = ok ? wy0 : 0.0f;
      ob[r * 4 + 2] = ok ? wx1 : 0.0f; ob[r * 4 + 3] = ok ? wy1 : 0.0f;
      os[r] = ok ? si : 0.0f;
      oc[r] = ok ? (float)(cinfo[wi] & 31u) : 0.0f;
    }
    if (ok) {
      for (int i = tid; i < K_PRE; i += 256) {
        if (cscore[i] > NEG_INF) {
          float iou = iou_box(wx0, wy0, wx1, wy1,
                              cbox[i][0], cbox[i][1], cbox[i][2], cbox[i][3]);
          if (iou > NMS_THR) cscore[i] = NEG_INF;   // self-IoU==1 removes winner
        }
      }
    }
    __syncthreads();
  }
}

// ---------------------------------------------------------------------------
extern "C" void kernel_launch(void* const* d_in, const int* in_sizes, int n_in,
                              void* d_out, int out_size, void* d_ws, size_t ws_size,
                              hipStream_t stream) {
  const float* anchors = (const float*)d_in[0];   // (A,4)
  const float* gt      = (const float*)d_in[1];   // (B,L,5)
  const float* y_pred  = (const float*)d_in[2];   // (B,A,25)
  float* out = (float*)d_out;

  unsigned long long* bestKey = (unsigned long long*)d_ws;
  float* lossAcc = (float*)((char*)d_ws + (size_t)BB * LL * sizeof(unsigned long long));

  float* det_boxes   = out + 1;
  float* det_scores  = det_boxes + (size_t)BB * MAX_DET * 4;
  float* det_classes = det_scores + (size_t)BB * MAX_DET;

  dim3 grid((AA + 255) / 256, BB);

  k_init<<<1, 256, 0, stream>>>(bestKey, lossAcc);
  k_best_anchor<<<grid, 256, 0, stream>>>(anchors, gt, bestKey);
  k_loss<<<grid, 256, 0, stream>>>(anchors, gt, y_pred, bestKey, lossAcc);
  k_finalize<<<1, 1, 0, stream>>>(lossAcc, out);
  k_nms<<<BB, 256, 0, stream>>>(anchors, y_pred, det_boxes, det_scores, det_classes);
}
